// RBFnet_30511447670975
// MI455X (gfx1250) — matrix operand order repaired and checked by reading, not compiled, not run on hardware
//
#include <hip/hip_runtime.h>

// ---------------------------------------------------------------------------
// Types for CDNA5 WMMA (wave32)
// ---------------------------------------------------------------------------
typedef __attribute__((ext_vector_type(16))) __bf16 v16bf;
typedef __attribute__((ext_vector_type(8)))  float  v8f;
typedef __attribute__((ext_vector_type(4)))  float  f32x4;
typedef __attribute__((ext_vector_type(4)))  __bf16 v4bf;

#define D      3072
#define NTRAIN 16384
#define NTEST  8192
#define NCLS   10
#define GAMMA_INV 0.1f

// LDS tile: 128 rows x 32 bf16 (64B payload) padded to 80B row stride so the
// 16 row-lanes of a fragment load hit 16 distinct banks (gcd(20,64)=4, period
// 16 -> conflict-free), while keeping 16B alignment for ds_load_b128.
#define LDS_ROW_BF 40   // elements (80 bytes)

__device__ __forceinline__ uint32_t lds_off(const void* p) {
  // For LDS, flat address low 32 bits == wave-relative LDS byte offset.
  return (uint32_t)(uintptr_t)p;
}

// Async global->LDS copy of one 16B chunk per lane (CDNA5, ASYNCcnt-tracked).
__device__ __forceinline__ void async_b128(uint32_t lds_byte, uint64_t gaddr) {
  asm volatile("global_load_async_to_lds_b128 %0, %1, off"
               :: "v"(lds_byte), "v"(gaddr) : "memory");
}
__device__ __forceinline__ void wait_async0() {
  asm volatile("s_wait_asynccnt 0x0" ::: "memory");
}

// Stage one 128x32 bf16 tile (gbase = row0 of tile, already offset by k0).
// 512 chunks of 16B; 2 per thread for a 256-thread block.
__device__ __forceinline__ void stage_tile_async(const __bf16* gbase,
                                                 uint32_t lds_base, int tid) {
#pragma unroll
  for (int i = 0; i < 2; ++i) {
    int t = tid + 256 * i;
    int row = t >> 2, ch = t & 3;
    uint64_t g = (uint64_t)(uintptr_t)(gbase + (size_t)row * D + ch * 8);
    async_b128(lds_base + row * 80 + ch * 16, g);
  }
}

// ---------------------------------------------------------------------------
// WMMA fragment loaders from LDS tiles (NT layout, see round-1 derivation).
// A frag (16x32): lanes 0-15 K[0..7]+K[16..23], lanes 16-31 K[8..15]+K[24..31]
// B frag (32x16 from row-major NT source): lanes 0-15 K[0..7]+K[16..23], lanes 16-31 K[8..15]+K[24..31]
// ---------------------------------------------------------------------------
__device__ __forceinline__ v16bf frag_a_lds(const __bf16* base, int row, int h) {
  v16bf v;
  ((f32x4*)&v)[0] = *(const f32x4*)(base + row * LDS_ROW_BF + 8 * h);
  ((f32x4*)&v)[1] = *(const f32x4*)(base + row * LDS_ROW_BF + 16 + 8 * h);
  return v;
}
__device__ __forceinline__ v16bf frag_b_lds(const __bf16* base, int row, int h) {
  v16bf v;
  ((f32x4*)&v)[0] = *(const f32x4*)(base + row * LDS_ROW_BF + 8 * h);
  ((f32x4*)&v)[1] = *(const f32x4*)(base + row * LDS_ROW_BF + 16 + 8 * h);
  return v;
}
__device__ __forceinline__ v8f wmma_bf16(v16bf a, v16bf b, v8f c) {
  return __builtin_amdgcn_wmma_f32_16x16x32_bf16(false, a, false, b,
                                                 (short)0, c, false, false);
}

// ---------------------------------------------------------------------------
// f32 -> bf16 convert (vectorized x4)
// ---------------------------------------------------------------------------
__global__ __launch_bounds__(256) void k_f32_to_bf16(const float* __restrict__ in,
                                                     __bf16* __restrict__ out, int n4) {
  int i = blockIdx.x * 256 + threadIdx.x;
  if (i >= n4) return;
  f32x4 v = ((const f32x4*)in)[i];
  v4bf o;
  o[0] = (__bf16)v[0]; o[1] = (__bf16)v[1];
  o[2] = (__bf16)v[2]; o[3] = (__bf16)v[3];
  ((v4bf*)out)[i] = o;
}

// ---------------------------------------------------------------------------
// Transpose+convert: Mt_bf[n*D + k] = (bf16) M[k*D + n]
// ---------------------------------------------------------------------------
__global__ __launch_bounds__(256) void k_transpose_bf16(const float* __restrict__ M,
                                                        __bf16* __restrict__ Mt) {
  __shared__ float t[32][33];
  int x = blockIdx.x * 32 + threadIdx.x;
  int y0 = blockIdx.y * 32 + threadIdx.y;
#pragma unroll
  for (int j = 0; j < 4; ++j)
    t[threadIdx.y + 8 * j][threadIdx.x] = M[(size_t)(y0 + 8 * j) * D + x];
  __syncthreads();
  int xo = blockIdx.y * 32 + threadIdx.x;
  int yo = blockIdx.x * 32 + threadIdx.y;
#pragma unroll
  for (int j = 0; j < 4; ++j)
    Mt[(size_t)(yo + 8 * j) * D + xo] = (__bf16)t[threadIdx.x][threadIdx.y + 8 * j];
}

// ---------------------------------------------------------------------------
// NT store-GEMM: C[i,n] = sum_k A[i,k]*B[n,k].  Block tile 128x128x(K=3072),
// 8 waves (4M x 2N), wave tile 32x64.  Operands staged to LDS with async
// double buffering; fragments read back via ds_load_b128.
// ---------------------------------------------------------------------------
template <bool STORE_BF>
__global__ __launch_bounds__(256) void k_gemm_nt_store(const __bf16* __restrict__ A,
                                                       const __bf16* __restrict__ B,
                                                       float* __restrict__ Cf,
                                                       __bf16* __restrict__ Cbf) {
  const int tid = threadIdx.x, lane = tid & 31, wid = tid >> 5;
  const int wm = wid & 3, wn = wid >> 2;
  const int m15 = lane & 15, h = lane >> 4;
  const int tileM = blockIdx.y * 128, tileN = blockIdx.x * 128;

  __shared__ __bf16 As[2][128 * LDS_ROW_BF];
  __shared__ __bf16 Bs[2][128 * LDS_ROW_BF];
  const uint32_t aB[2] = {lds_off(&As[0][0]), lds_off(&As[1][0])};
  const uint32_t bB[2] = {lds_off(&Bs[0][0]), lds_off(&Bs[1][0])};

  const __bf16* Ag = A + (size_t)tileM * D;
  const __bf16* Bg = B + (size_t)tileN * D;
  const int arow[2] = {32 * wm + 16 * 0 + m15, 32 * wm + 16 * 1 + m15};
  const int brow[4] = {64 * wn + m15, 64 * wn + 16 + m15,
                       64 * wn + 32 + m15, 64 * wn + 48 + m15};

  v8f acc[2][4];
#pragma unroll
  for (int mt = 0; mt < 2; ++mt)
#pragma unroll
    for (int nt = 0; nt < 4; ++nt) acc[mt][nt] = (v8f){0.f,0.f,0.f,0.f,0.f,0.f,0.f,0.f};

  stage_tile_async(Ag, aB[0], tid);
  stage_tile_async(Bg, bB[0], tid);
  wait_async0();
  __syncthreads();

  int cur = 0;
  for (int k0 = 0; k0 < D; k0 += 32) {
    const int nxt = cur ^ 1;
    if (k0 + 32 < D) {
      stage_tile_async(Ag + k0 + 32, aB[nxt], tid);
      stage_tile_async(Bg + k0 + 32, bB[nxt], tid);
    }
    v16bf a[2], b[4];
#pragma unroll
    for (int mt = 0; mt < 2; ++mt) a[mt] = frag_a_lds(&As[cur][0], arow[mt], h);
#pragma unroll
    for (int nt = 0; nt < 4; ++nt) b[nt] = frag_b_lds(&Bs[cur][0], brow[nt], h);
#pragma unroll
    for (int mt = 0; mt < 2; ++mt)
#pragma unroll
      for (int nt = 0; nt < 4; ++nt) acc[mt][nt] = wmma_bf16(a[mt], b[nt], acc[mt][nt]);
    wait_async0();      // next tiles resident
    __syncthreads();    // all waves done reading `cur` before it is re-staged
    cur = nxt;
  }

#pragma unroll
  for (int mt = 0; mt < 2; ++mt)
#pragma unroll
    for (int nt = 0; nt < 4; ++nt)
#pragma unroll
      for (int r = 0; r < 8; ++r) {
        int gi = tileM + 32 * wm + 16 * mt + 8 * h + r;
        int gn = tileN + 64 * wn + 16 * nt + m15;
        float v = acc[mt][nt][r];
        Cf[(size_t)gi * D + gn] = v;
        if constexpr (STORE_BF) Cbf[(size_t)gi * D + gn] = (__bf16)v;
      }
}

// ---------------------------------------------------------------------------
// Row dot: out[row] = sum_k a[row,k]*b[row,k]  (k = 3072)
// ---------------------------------------------------------------------------
__global__ __launch_bounds__(256) void k_rowdot(const float* __restrict__ a,
                                                const float* __restrict__ b,
                                                float* __restrict__ out) {
  int row = blockIdx.x;
  const f32x4* av = (const f32x4*)(a + (size_t)row * D);
  const f32x4* bv = (const f32x4*)(b + (size_t)row * D);
  float s = 0.f;
  for (int t = threadIdx.x; t < D / 4; t += 256) {
    f32x4 x = av[t], y = bv[t];
    s += x[0]*y[0] + x[1]*y[1] + x[2]*y[2] + x[3]*y[3];
  }
#pragma unroll
  for (int o = 16; o > 0; o >>= 1) s += __shfl_xor(s, o);
  __shared__ float red[8];
  if ((threadIdx.x & 31) == 0) red[threadIdx.x >> 5] = s;
  __syncthreads();
  if (threadIdx.x == 0) {
    float t2 = 0.f;
#pragma unroll
    for (int i = 0; i < 8; ++i) t2 += red[i];
    out[row] = t2;
  }
}

__global__ __launch_bounds__(256) void k_zero(float* __restrict__ p, int n) {
  int i = blockIdx.x * 256 + threadIdx.x;
  if (i < n) p[i] = 0.f;
}

// ---------------------------------------------------------------------------
// Fused GEMM-3 + Laplacian RBF + reduction over train rows.
// P[i,j] = sum_k sM_bf[i,k]*xte_bf[j,k]; d2 = sn[i]+cn[j]-2P;
// K = exp(-sqrt(max(d2,0))/10); sumK[j] += K; accY[j,c] += K*y[i,c].
// Grid: (NTEST/128 j-blocks, 16 i-splits); each block loops over 8 i-tiles.
// ---------------------------------------------------------------------------
#define ITILES_PER_BLOCK 8
__global__ __launch_bounds__(256) void k_gemm_rbf_fused(
    const __bf16* __restrict__ A,        // sM_bf  [NTRAIN x D]
    const __bf16* __restrict__ B,        // xte_bf [NTEST x D]
    const float* __restrict__ s_norm2,   // [NTRAIN]
    const float* __restrict__ c_norm2,   // [NTEST]
    const float* __restrict__ y_train,   // [NTRAIN x 10]
    float* __restrict__ accY,            // [NTEST x 10]
    float* __restrict__ sumK) {          // [NTEST]
  const int tid = threadIdx.x, lane = tid & 31, wid = tid >> 5;
  const int wm = wid & 3, wn = wid >> 2;
  const int m15 = lane & 15, h = lane >> 4;
  const int tileN = blockIdx.x * 128;

  __shared__ __bf16 As[2][128 * LDS_ROW_BF];
  __shared__ __bf16 Bs[2][128 * LDS_ROW_BF];
  __shared__ float ldsY[128 * 12];    // y_train tile, stride 12 (16B aligned)
  __shared__ float ldsAcc[128 * 11];  // per-j: c=0..9 sums, c=10 = sumK
  const uint32_t aB[2] = {lds_off(&As[0][0]), lds_off(&As[1][0])};
  const uint32_t bB[2] = {lds_off(&Bs[0][0]), lds_off(&Bs[1][0])};

  for (int idx = tid; idx < 128 * 11; idx += 256) ldsAcc[idx] = 0.f;

  const __bf16* Bg = B + (size_t)tileN * D;
  const int arow[2] = {32 * wm + 16 * 0 + m15, 32 * wm + 16 * 1 + m15};
  const int brow[4] = {64 * wn + m15, 64 * wn + 16 + m15,
                       64 * wn + 32 + m15, 64 * wn + 48 + m15};
  float cnv[4];
#pragma unroll
  for (int nt = 0; nt < 4; ++nt)
    cnv[nt] = c_norm2[tileN + 64 * wn + 16 * nt + m15];

  for (int it = 0; it < ITILES_PER_BLOCK; ++it) {
    const int tileM = (blockIdx.y * ITILES_PER_BLOCK + it) * 128;
    const __bf16* Ag = A + (size_t)tileM * D;

    __syncthreads();  // previous epilogue done with ldsY
    for (int idx = tid; idx < 128 * NCLS; idx += 256) {
      int rl = idx / NCLS, c = idx % NCLS;
      ldsY[rl * 12 + c] = y_train[(size_t)(tileM + rl) * NCLS + c];
    }
    stage_tile_async(Ag, aB[0], tid);
    stage_tile_async(Bg, bB[0], tid);
    wait_async0();
    __syncthreads();

    v8f acc[2][4];
#pragma unroll
    for (int mt = 0; mt < 2; ++mt)
#pragma unroll
      for (int nt = 0; nt < 4; ++nt) acc[mt][nt] = (v8f){0.f,0.f,0.f,0.f,0.f,0.f,0.f,0.f};

    int cur = 0;
    for (int k0 = 0; k0 < D; k0 += 32) {
      const int nxt = cur ^ 1;
      if (k0 + 32 < D) {
        stage_tile_async(Ag + k0 + 32, aB[nxt], tid);
        stage_tile_async(Bg + k0 + 32, bB[nxt], tid);
      }
      v16bf a[2], b[4];
#pragma unroll
      for (int mt = 0; mt < 2; ++mt) a[mt] = frag_a_lds(&As[cur][0], arow[mt], h);
#pragma unroll
      for (int nt = 0; nt < 4; ++nt) b[nt] = frag_b_lds(&Bs[cur][0], brow[nt], h);
#pragma unroll
      for (int mt = 0; mt < 2; ++mt)
#pragma unroll
        for (int nt = 0; nt < 4; ++nt) acc[mt][nt] = wmma_bf16(a[mt], b[nt], acc[mt][nt]);
      wait_async0();
      __syncthreads();
      cur = nxt;
    }

    // Epilogue: scalar RBF math + reduction over this i-tile's 128 rows.
#pragma unroll
    for (int nt = 0; nt < 4; ++nt) {
      const float cn = cnv[nt];
#pragma unroll
      for (int mt = 0; mt < 2; ++mt) {
        const int rl0 = 32 * wm + 16 * mt + 8 * h;
        float snA[8];
        *(f32x4*)snA       = *(const f32x4*)(s_norm2 + tileM + rl0);
        *((f32x4*)snA + 1) = *(const f32x4*)(s_norm2 + tileM + rl0 + 4);
        float kS = 0.f, yS[NCLS];
#pragma unroll
        for (int c = 0; c < NCLS; ++c) yS[c] = 0.f;
#pragma unroll
        for (int r = 0; r < 8; ++r) {
          float P = acc[mt][nt][r];
          float d2 = fmaf(-2.f, P, snA[r] + cn);
          float dist = __builtin_sqrtf(fmaxf(d2, 0.f));
          float K = __expf(dist * (-GAMMA_INV));
          kS += K;
          const float* yp = &ldsY[(rl0 + r) * 12];
#pragma unroll
          for (int c = 0; c < NCLS; ++c) yS[c] += K * yp[c];
        }
        kS += __shfl_xor(kS, 16);
#pragma unroll
        for (int c = 0; c < NCLS; ++c) yS[c] += __shfl_xor(yS[c], 16);
        if (h == 0) {
          int jl = 64 * wn + 16 * nt + m15;
          atomicAdd(&ldsAcc[jl * 11 + 10], kS);
#pragma unroll
          for (int c = 0; c < NCLS; ++c) atomicAdd(&ldsAcc[jl * 11 + c], yS[c]);
        }
      }
    }
  }

  __syncthreads();
  for (int idx = tid; idx < 128 * 11; idx += 256) {
    int jl = idx / 11, c = idx % 11;
    float v = ldsAcc[idx];
    int jg = tileN + jl;
    if (c == 10) atomicAdd(&sumK[jg], v);
    else         atomicAdd(&accY[(size_t)jg * NCLS + c], v);
  }
}

__global__ __launch_bounds__(256) void k_finalize(const float* __restrict__ accY,
                                                  const float* __restrict__ sumK,
                                                  float* __restrict__ out, int n) {
  int i = blockIdx.x * 256 + threadIdx.x;
  if (i < n) out[i] = accY[i] / sumK[i / NCLS];
}

// ---------------------------------------------------------------------------
// Host launcher
// ---------------------------------------------------------------------------
extern "C" void kernel_launch(void* const* d_in, const int* in_sizes, int n_in,
                              void* d_out, int out_size, void* d_ws, size_t ws_size,
                              hipStream_t stream) {
  const float* x_train = (const float*)d_in[0];
  const float* y_train = (const float*)d_in[1];
  const float* x_test  = (const float*)d_in[2];
  const float* M       = (const float*)d_in[3];

  char* ws = (char*)d_ws;
  auto carve = [&](size_t bytes) { char* p = ws; ws += (bytes + 255) & ~size_t(255); return p; };
  __bf16* Mt_bf   = (__bf16*)carve((size_t)D * D * 2);
  __bf16* xtr_bf  = (__bf16*)carve((size_t)NTRAIN * D * 2);
  __bf16* xte_bf  = (__bf16*)carve((size_t)NTEST * D * 2);
  float*  sM_f32  = (float*)carve((size_t)NTRAIN * D * 4);
  __bf16* sM_bf   = (__bf16*)carve((size_t)NTRAIN * D * 2);
  float*  cM_f32  = (float*)carve((size_t)NTEST * D * 4);
  float*  s_norm2 = (float*)carve((size_t)NTRAIN * 4);
  float*  c_norm2 = (float*)carve((size_t)NTEST * 4);
  float*  sumK    = (float*)carve((size_t)NTEST * 4);
  float*  accY    = (float*)carve((size_t)NTEST * NCLS * 4);

  // 1) bf16 conversions + M transpose
  {
    int n4 = NTRAIN * D / 4;
    k_f32_to_bf16<<<(n4 + 255) / 256, 256, 0, stream>>>(x_train, xtr_bf, n4);
    n4 = NTEST * D / 4;
    k_f32_to_bf16<<<(n4 + 255) / 256, 256, 0, stream>>>(x_test, xte_bf, n4);
    k_transpose_bf16<<<dim3(D / 32, D / 32), dim3(32, 8), 0, stream>>>(M, Mt_bf);
  }
  // 2) sM = x_train @ M (NT vs Mt): f32 + bf16 outputs
  k_gemm_nt_store<true><<<dim3(D / 128, NTRAIN / 128), 256, 0, stream>>>(xtr_bf, Mt_bf, sM_f32, sM_bf);
  // 3) cM = x_test @ M: f32 only
  k_gemm_nt_store<false><<<dim3(D / 128, NTEST / 128), 256, 0, stream>>>(xte_bf, Mt_bf, cM_f32, (__bf16*)nullptr);
  // 4) Mahalanobis norms
  k_rowdot<<<NTRAIN, 256, 0, stream>>>(sM_f32, x_train, s_norm2);
  k_rowdot<<<NTEST, 256, 0, stream>>>(cM_f32, x_test, c_norm2);
  // 5) zero accumulators
  k_zero<<<(NTEST + 255) / 256, 256, 0, stream>>>(sumK, NTEST);
  k_zero<<<(NTEST * NCLS + 255) / 256, 256, 0, stream>>>(accY, NTEST * NCLS);
  // 6) fused GEMM + RBF + reduction
  k_gemm_rbf_fused<<<dim3(NTEST / 128, NTRAIN / 128 / ITILES_PER_BLOCK), 256, 0, stream>>>(
      sM_bf, xte_bf, s_norm2, c_norm2, y_train, accY, sumK);
  // 7) normalize -> tuple output 0
  k_finalize<<<(NTEST * NCLS + 255) / 256, 256, 0, stream>>>(accY, sumK, (float*)d_out, NTEST * NCLS);
  // 8) tuple output 1: M passthrough
  hipMemcpyAsync((float*)d_out + (size_t)NTEST * NCLS, M, (size_t)D * D * 4,
                 hipMemcpyDeviceToDevice, stream);
}
